// TtCausalSelfAttention_19885698580998
// MI455X (gfx1250) — matrix operand order repaired and checked by reading, not compiled, not run on hardware
//
#include <hip/hip_runtime.h>
#include <hip/hip_bf16.h>

// ---------------------------------------------------------------------------
// CDNA5 (gfx1250) causal self-attention:
//   - bf16 WMMA (16x16x32, f32 accum) for GEMMs and flash attention
//   - async global->LDS staging (ASYNCcnt), double-buffered, in GEMM and flash
//   - GEMM wave tile 64x64 (16 WMMAs per K-step per wave, 1.0 LDS-load/WMMA)
//   - 4-wave flash blocks share K/V tiles staged once per block
//   - DPP16 butterfly reductions (VALU-only) in flash softmax
// ---------------------------------------------------------------------------

typedef __attribute__((ext_vector_type(16))) __bf16 v16bf;
typedef __attribute__((ext_vector_type(8)))  float  v8f;

union Frag16 { uint4 u[2]; v16bf v; };
union Frag8  { uint4 u;    __bf16 h[8]; };

__device__ __forceinline__ v16bf ld_pair(const __bf16* p0, const __bf16* p1) {
    Frag16 f;
    f.u[0] = *reinterpret_cast<const uint4*>(p0);
    f.u[1] = *reinterpret_cast<const uint4*>(p1);
    return f.v;
}

__device__ __forceinline__ v8f wmma_bf16(v16bf a, v16bf b, v8f c) {
    return __builtin_amdgcn_wmma_f32_16x16x32_bf16(false, a, false, b,
                                                   (short)0, c, false, false);
}

__device__ __forceinline__ v8f vzero8() {
    v8f z = {0.f, 0.f, 0.f, 0.f, 0.f, 0.f, 0.f, 0.f};
    return z;
}

// --- CDNA5 async global->LDS copy (tracked by ASYNCcnt) ---------------------
__device__ __forceinline__ void async_ld_b128(const void* gptr, void* lptr) {
    uint64_t g = (uint64_t)(uintptr_t)gptr;            // global address
    uint32_t l = (uint32_t)(uintptr_t)lptr;            // low 32b = LDS offset
    asm volatile("global_load_async_to_lds_b128 %0, %1, off"
                 :: "v"(l), "v"(g) : "memory");
}
__device__ __forceinline__ void wait_async_le6() {
    asm volatile("s_wait_asynccnt 6" ::: "memory");
}
__device__ __forceinline__ void wait_async_le2() {
    asm volatile("s_wait_asynccnt 2" ::: "memory");
}
__device__ __forceinline__ void wait_async_0() {
    asm volatile("s_wait_asynccnt 0" ::: "memory");
}

// --- DPP16 butterfly reduction over 16-lane groups (VALU only, no LDS) ------
template <int CTRL>
__device__ __forceinline__ float dpp_mov(float x) {
    return __int_as_float(__builtin_amdgcn_update_dpp(
        0, __float_as_int(x), CTRL, 0xF, 0xF, true));
}
__device__ __forceinline__ float red_max16(float x) {
    x = fmaxf(x, dpp_mov<0xB1>(x));    // quad_perm [1,0,3,2]  : xor 1
    x = fmaxf(x, dpp_mov<0x4E>(x));    // quad_perm [2,3,0,1]  : xor 2
    x = fmaxf(x, dpp_mov<0x141>(x));   // row_half_mirror      : xor 7
    x = fmaxf(x, dpp_mov<0x140>(x));   // row_mirror           : xor 15
    return x;
}
__device__ __forceinline__ float red_add16(float x) {
    x += dpp_mov<0xB1>(x);
    x += dpp_mov<0x4E>(x);
    x += dpp_mov<0x141>(x);
    x += dpp_mov<0x140>(x);
    return x;
}

// ---------------------------------------------------------------------------
// fp32 -> bf16 conversion
// ---------------------------------------------------------------------------
__global__ __launch_bounds__(256)
void cvt_f32_bf16(const float* __restrict__ in, __bf16* __restrict__ out, int n) {
    int i = (blockIdx.x * 256 + threadIdx.x) * 4;
    if (i + 3 < n) {
        float4 f = *reinterpret_cast<const float4*>(in + i);
        out[i + 0] = (__bf16)f.x;
        out[i + 1] = (__bf16)f.y;
        out[i + 2] = (__bf16)f.z;
        out[i + 3] = (__bf16)f.w;
    }
}

// ---------------------------------------------------------------------------
// C[M,N] = A[M,K] * B[N,K]^T + bias[N]  ("NT", both K-contiguous)
// Block tile 128(M) x 256(N), 8 waves (2x4), wave tile 64x64 = 4x4 WMMA frags.
// K stepped by 32 with double-buffered async global->LDS staging.
// ---------------------------------------------------------------------------
template <typename OutT>
__global__ __launch_bounds__(256)
void gemm_bf16_nt(const __bf16* __restrict__ A,
                  const __bf16* __restrict__ Bm,
                  const float*  __restrict__ bias,
                  OutT*         __restrict__ Cmat,
                  int M, int N, int K) {
    constexpr int LDT = 40;  // 32 + 8 pad; rows 80B (16B aligned)
    __shared__ __align__(16) __bf16 As[2][128 * LDT];
    __shared__ __align__(16) __bf16 Bs[2][256 * LDT];

    const int tid   = threadIdx.x;
    const int lane  = tid & 31;
    const int wid   = tid >> 5;
    const int waveM = wid >> 2;   // 0..1
    const int waveN = wid & 3;    // 0..3
    const int bm    = blockIdx.y * 128;
    const int bn    = blockIdx.x * 256;
    const int lm    = lane & 15;
    const int hi    = lane >> 4;

    const int srow  = tid >> 2;           // 0..63
    const int scc   = (tid & 3) * 8;      // 0,8,16,24
    const __bf16* Arow0 = A  + (size_t)(bm + srow)       * K + scc;
    const __bf16* Arow1 = A  + (size_t)(bm + srow + 64)  * K + scc;
    const __bf16* Brow0 = Bm + (size_t)(bn + srow)       * K + scc;
    const __bf16* Brow1 = Bm + (size_t)(bn + srow + 64)  * K + scc;
    const __bf16* Brow2 = Bm + (size_t)(bn + srow + 128) * K + scc;
    const __bf16* Brow3 = Bm + (size_t)(bn + srow + 192) * K + scc;

    v8f acc[4][4];
#pragma unroll
    for (int i = 0; i < 4; ++i)
#pragma unroll
        for (int j = 0; j < 4; ++j) acc[i][j] = vzero8();

    // prologue: stage tile 0 (6 async b128 per thread)
    async_ld_b128(Arow0, &As[0][srow * LDT + scc]);
    async_ld_b128(Arow1, &As[0][(srow + 64) * LDT + scc]);
    async_ld_b128(Brow0, &Bs[0][srow * LDT + scc]);
    async_ld_b128(Brow1, &Bs[0][(srow + 64) * LDT + scc]);
    async_ld_b128(Brow2, &Bs[0][(srow + 128) * LDT + scc]);
    async_ld_b128(Brow3, &Bs[0][(srow + 192) * LDT + scc]);

    const int nk = K >> 5;
    for (int i = 0; i < nk; ++i) {
        const int cur = i & 1;
        if (i + 1 < nk) {
            const int k1 = (i + 1) * 32;
            async_ld_b128(Arow0 + k1, &As[cur ^ 1][srow * LDT + scc]);
            async_ld_b128(Arow1 + k1, &As[cur ^ 1][(srow + 64) * LDT + scc]);
            async_ld_b128(Brow0 + k1, &Bs[cur ^ 1][srow * LDT + scc]);
            async_ld_b128(Brow1 + k1, &Bs[cur ^ 1][(srow + 64) * LDT + scc]);
            async_ld_b128(Brow2 + k1, &Bs[cur ^ 1][(srow + 128) * LDT + scc]);
            async_ld_b128(Brow3 + k1, &Bs[cur ^ 1][(srow + 192) * LDT + scc]);
            wait_async_le6();   // tile i resident; tile i+1 in flight
        } else {
            wait_async_0();
        }
        __syncthreads();

        v16bf af[4], bfr[4];
#pragma unroll
        for (int fm = 0; fm < 4; ++fm) {
            const __bf16* p = &As[cur][(waveM * 64 + fm * 16 + lm) * LDT + hi * 8];
            af[fm] = ld_pair(p, p + 16);
        }
#pragma unroll
        for (int fn = 0; fn < 4; ++fn) {
            const __bf16* p = &Bs[cur][(waveN * 64 + fn * 16 + lm) * LDT + hi * 8];
            bfr[fn] = ld_pair(p, p + 16);
        }
#pragma unroll
        for (int fm = 0; fm < 4; ++fm)
#pragma unroll
            for (int fn = 0; fn < 4; ++fn)
                acc[fm][fn] = wmma_bf16(af[fm], bfr[fn], acc[fm][fn]);
        __syncthreads();
    }

#pragma unroll
    for (int fm = 0; fm < 4; ++fm)
#pragma unroll
        for (int fn = 0; fn < 4; ++fn) {
            int   gn = bn + waveN * 64 + fn * 16 + lm;
            float bv = bias[gn];
#pragma unroll
            for (int r = 0; r < 8; ++r) {
                int gm = bm + waveM * 64 + fm * 16 + r + hi * 8;
                Cmat[(size_t)gm * N + gn] = (OutT)(acc[fm][fn][r] + bv);
            }
        }
}

// ---------------------------------------------------------------------------
// Flash attention: 4 waves per block, 128 q rows per block (32 per wave).
// K tiles staged once per block via async LDS loads (double buffered);
// V tiles transposed into LDS cooperatively. Wave w consumes key tiles
// j <= qb4+w (exact causal alignment); only the diagonal tile is masked.
// ---------------------------------------------------------------------------
__global__ __launch_bounds__(128)
void flash_attn_kernel(const __bf16* __restrict__ qkv, __bf16* __restrict__ Y) {
    constexpr int T = 2048, C = 1024, Dh = 64, S3 = 3 * C;
    constexpr int LDV = 40;   // Vt/Ps row stride (80B)
    constexpr int LDK = 72;   // Ks row stride (144B)
    __shared__ __align__(16) __bf16 Ks[2][32 * LDK];   // raw K rows [k][d]
    __shared__ __align__(16) __bf16 Vt[64 * LDV];      // V^T [d][k]
    __shared__ __align__(16) __bf16 Ps[4][32 * LDV];   // per-wave P tiles

    const int tid  = threadIdx.x;
    const int lane = tid & 31;
    const int w    = tid >> 5;
    const int lm   = lane & 15;
    const int hi   = lane >> 4;
    const int qb4  = blockIdx.x * 4;
    const int b    = blockIdx.y >> 4;
    const int h    = blockIdx.y & 15;
    const int qw   = qb4 + w;          // wave's 32-row q block == its last tile

    const __bf16* base = qkv + (size_t)b * T * S3 + h * Dh;
    const __bf16* qp = base;
    const __bf16* kp = base + C;
    const __bf16* vp = base + 2 * C;

    // staging coordinates: 2 chunks (8 bf16) per thread covering 32x64 tile
    const int r0 = tid >> 3;           // rows 0..15
    const int r1 = r0 + 16;            // rows 16..31
    const int dc = (tid & 7) * 8;      // d chunk

    // Q fragments: 2 subtiles x 2 d-halves (A layout)
    v16bf qf[2][2];
#pragma unroll
    for (int t = 0; t < 2; ++t) {
        const __bf16* qr = qp + (size_t)(qw * 32 + t * 16 + lm) * S3;
#pragma unroll
        for (int f = 0; f < 2; ++f) {
            const __bf16* p = qr + f * 32 + hi * 8;
            qf[t][f] = ld_pair(p, p + 16);
        }
    }

    v8f o[2][4];
#pragma unroll
    for (int t = 0; t < 2; ++t)
#pragma unroll
        for (int f = 0; f < 4; ++f) o[t][f] = vzero8();
    float mi[2][8], li[2][8];
#pragma unroll
    for (int t = 0; t < 2; ++t)
#pragma unroll
        for (int r = 0; r < 8; ++r) { mi[t][r] = -1e30f; li[t][r] = 0.f; }

    // prologue: async stage K tile 0
    async_ld_b128(kp + (size_t)r0 * S3 + dc, &Ks[0][r0 * LDK + dc]);
    async_ld_b128(kp + (size_t)r1 * S3 + dc, &Ks[0][r1 * LDK + dc]);

    const int jendb = qb4 + 3;
    for (int j = 0; j <= jendb; ++j) {
        const int kbase = j * 32;
        const int cur   = j & 1;

        // --- V tile: global -> regs -> transposed scatter (shared by block)
        Frag8 u0, u1;
        u0.u = *reinterpret_cast<const uint4*>(vp + (size_t)(kbase + r0) * S3 + dc);
        u1.u = *reinterpret_cast<const uint4*>(vp + (size_t)(kbase + r1) * S3 + dc);
#pragma unroll
        for (int e = 0; e < 8; ++e) Vt[(dc + e) * LDV + r0] = u0.h[e];
#pragma unroll
        for (int e = 0; e < 8; ++e) Vt[(dc + e) * LDV + r1] = u1.h[e];

        // --- prefetch next K tile (async, double buffered)
        if (j < jendb) {
            const int nb = kbase + 32;
            async_ld_b128(kp + (size_t)(nb + r0) * S3 + dc, &Ks[cur ^ 1][r0 * LDK + dc]);
            async_ld_b128(kp + (size_t)(nb + r1) * S3 + dc, &Ks[cur ^ 1][r1 * LDK + dc]);
            wait_async_le2();       // tile j resident, tile j+1 in flight
        } else {
            wait_async_0();
        }
        __syncthreads();            // K tile j + V^T tile visible to all waves

        if (j <= qw) {              // wave-uniform: EXEC stays all-ones inside
            // K fragments from LDS (B layout), shared by both subtiles
            v16bf kf[2][2];
#pragma unroll
            for (int s = 0; s < 2; ++s) {
                const __bf16* krow = &Ks[cur][(s * 16 + lm) * LDK];
#pragma unroll
                for (int ds = 0; ds < 2; ++ds) {
                    const __bf16* p = krow + ds * 32 + hi * 8;
                    kf[s][ds] = ld_pair(p, p + 16);
                }
            }

            const bool diag = (j == qw);
            const int  c0 = kbase + lm;
            const int  c1 = kbase + 16 + lm;
#pragma unroll
            for (int t = 0; t < 2; ++t) {
                v8f s0 = vzero8(), s1 = vzero8();
                s0 = wmma_bf16(qf[t][0], kf[0][0], s0);
                s0 = wmma_bf16(qf[t][1], kf[0][1], s0);
                s1 = wmma_bf16(qf[t][0], kf[1][0], s1);
                s1 = wmma_bf16(qf[t][1], kf[1][1], s1);

                float pr0[8], pr1[8], corr[8];
#pragma unroll
                for (int r = 0; r < 8; ++r) {
                    float a  = s0[r] * 0.125f;       // 1/sqrt(64)
                    float bb = s1[r] * 0.125f;
                    if (diag) {                      // only diagonal tile masks
                        int tq = qw * 32 + t * 16 + hi * 8 + r;
                        if (c0 > tq) a  = -1e30f;
                        if (c1 > tq) bb = -1e30f;
                    }
                    float rm   = red_max16(fmaxf(a, bb));
                    float mnew = fmaxf(mi[t][r], rm);
                    float cr   = __expf(mi[t][r] - mnew);
                    mi[t][r] = mnew;
                    float p0 = __expf(a - mnew);
                    float p1 = __expf(bb - mnew);
                    float ps = red_add16(p0 + p1);
                    li[t][r] = li[t][r] * cr + ps;
                    corr[r] = cr;
                    pr0[r] = p0; pr1[r] = p1;
                }
#pragma unroll
                for (int f = 0; f < 4; ++f)
#pragma unroll
                    for (int r = 0; r < 8; ++r) o[t][f][r] *= corr[r];

                // P -> per-wave LDS region (C layout -> A layout bounce)
#pragma unroll
                for (int r = 0; r < 8; ++r) {
                    Ps[w][(t * 16 + hi * 8 + r) * LDV + lm]      = (__bf16)pr0[r];
                    Ps[w][(t * 16 + hi * 8 + r) * LDV + 16 + lm] = (__bf16)pr1[r];
                }
            }

            // same-wave LDS RAW: hardware keeps DS in order; compiler waits
            v16bf pf[2];
#pragma unroll
            for (int t = 0; t < 2; ++t) {
                const __bf16* p = &Ps[w][(t * 16 + lm) * LDV + hi * 8];
                pf[t] = ld_pair(p, p + 16);
            }
#pragma unroll
            for (int f = 0; f < 4; ++f) {
                const __bf16* p = &Vt[(f * 16 + lm) * LDV + hi * 8];
                v16bf vfr = ld_pair(p, p + 16);
#pragma unroll
                for (int t = 0; t < 2; ++t)
                    o[t][f] = wmma_bf16(pf[t], vfr, o[t][f]);
            }
        }
        __syncthreads();            // all reads done before tiles overwritten
    }

    // --- normalize + store y
    __bf16* yb = Y + (size_t)b * T * C + h * Dh;
#pragma unroll
    for (int t = 0; t < 2; ++t) {
        float inv[8];
#pragma unroll
        for (int r = 0; r < 8; ++r) inv[r] = 1.0f / li[t][r];
#pragma unroll
        for (int f = 0; f < 4; ++f)
#pragma unroll
            for (int r = 0; r < 8; ++r) {
                int tq = qw * 32 + t * 16 + hi * 8 + r;
                yb[(size_t)tq * C + f * 16 + lm] = (__bf16)(o[t][f][r] * inv[r]);
            }
    }
}

// ---------------------------------------------------------------------------
// Launch
// ---------------------------------------------------------------------------
extern "C" void kernel_launch(void* const* d_in, const int* in_sizes, int n_in,
                              void* d_out, int out_size, void* d_ws, size_t ws_size,
                              hipStream_t stream) {
    (void)in_sizes; (void)n_in; (void)out_size; (void)ws_size;
    const float* x      = (const float*)d_in[0];
    const float* w_attn = (const float*)d_in[1];
    const float* b_attn = (const float*)d_in[2];
    const float* w_proj = (const float*)d_in[3];
    const float* b_proj = (const float*)d_in[4];

    constexpr int Bb = 4, T = 2048, C = 1024;
    constexpr int M = Bb * T;
    constexpr size_t nx   = (size_t)M * C;
    constexpr size_t nwa  = (size_t)3 * C * C;
    constexpr size_t nwp  = (size_t)C * C;
    constexpr size_t nqkv = (size_t)M * 3 * C;

    auto align256 = [](size_t b) { return (b + 255) & ~(size_t)255; };
    char* w = (char*)d_ws;
    __bf16* xb  = (__bf16*)w; w += align256(nx * 2);
    __bf16* wab = (__bf16*)w; w += align256(nwa * 2);
    __bf16* wpb = (__bf16*)w; w += align256(nwp * 2);
    __bf16* qkv = (__bf16*)w; w += align256(nqkv * 2);
    __bf16* yb  = (__bf16*)w; w += align256(nx * 2);

    cvt_f32_bf16<<<(int)(nx  / 1024), 256, 0, stream>>>(x,      xb,  (int)nx);
    cvt_f32_bf16<<<(int)(nwa / 1024), 256, 0, stream>>>(w_attn, wab, (int)nwa);
    cvt_f32_bf16<<<(int)(nwp / 1024), 256, 0, stream>>>(w_proj, wpb, (int)nwp);

    gemm_bf16_nt<__bf16><<<dim3(3 * C / 256, M / 128), 256, 0, stream>>>(
        xb, wab, b_attn, qkv, M, 3 * C, C);

    flash_attn_kernel<<<dim3(T / 128, Bb * 16), 128, 0, stream>>>(qkv, yb);

    gemm_bf16_nt<float><<<dim3(C / 256, M / 128), 256, 0, stream>>>(
        yb, wpb, b_proj, (float*)d_out, M, C, C);
}
